// FaceFeatsModel_5531917877921
// MI455X (gfx1250) — compile-verified, matrix operand order checked by reading, not run on hardware
//
#include <hip/hip_runtime.h>
#include <hip/hip_bf16.h>

typedef __attribute__((ext_vector_type(16))) __bf16 v16bf;
typedef __attribute__((ext_vector_type(8)))  __bf16 v8bf;
typedef __attribute__((ext_vector_type(8)))  float  v8f;
typedef __attribute__((ext_vector_type(4)))  int    v4i;

// pointer-to-global-v4i / pointer-to-LDS-v4i, matching the async builtin's params
typedef __attribute__((address_space(1))) v4i* gv4i_p;
typedef __attribute__((address_space(3))) v4i* lv4i_p;

#define B_Q     2048
#define N_F     100000
#define D_F     512
#define NTILES  (N_F / 16)        // 6250
#define NSPLIT  50
#define TPB_T   (NTILES / NSPLIT) // 125 tiles per block
#define LDS_ROW 1040              // 1024 B row + 16 B pad -> bank-conflict-free ds_load_b128
#define BUF_SZ  (16 * LDS_ROW)    // one 16-face tile

#if defined(__has_builtin)
#if __has_builtin(__builtin_amdgcn_global_load_async_to_lds_b128) && \
    __has_builtin(__builtin_amdgcn_s_wait_asynccnt)
#define ASYNC_STAGE 1
#endif
#endif
#ifndef ASYNC_STAGE
#define ASYNC_STAGE 0
#endif

// ---------------------------------------------------------------------------
// Kernel 1: bf16-convert Q, zero-init packed argmax accumulators
// ---------------------------------------------------------------------------
__global__ void prep_q(const float* __restrict__ q, __bf16* __restrict__ qb,
                       unsigned long long* __restrict__ best) {
    int i = blockIdx.x * 256 + threadIdx.x;
    if (i < B_Q * D_F) qb[i] = (__bf16)q[i];
    if (i < B_Q)       best[i] = 0ull;   // minimum orderable key
}

// ---------------------------------------------------------------------------
// Kernel 2: per-row L2 norm of face_feats; emit pre-normalized bf16 copy
// ---------------------------------------------------------------------------
__global__ void norm_rows(const float* __restrict__ ff, __bf16* __restrict__ ffb,
                          float* __restrict__ inv_norm) {
    const int row = blockIdx.x;
    const int t   = threadIdx.x;          // 256 threads, 2 floats each
    const float2 v = ((const float2*)(ff + (size_t)row * D_F))[t];
    float ss = v.x * v.x + v.y * v.y;
    #pragma unroll
    for (int m = 16; m >= 1; m >>= 1) ss += __shfl_xor(ss, m, 32);   // wave32 reduce
    __shared__ float wsum[8];
    if ((t & 31) == 0) wsum[t >> 5] = ss;
    __syncthreads();
    float tot = 0.f;
    #pragma unroll
    for (int i = 0; i < 8; ++i) tot += wsum[i];
    const float inv = rsqrtf(tot);
    if (t == 0) inv_norm[row] = inv;
    __bf16* dst = ffb + (size_t)row * D_F + t * 2;
    dst[0] = (__bf16)(v.x * inv);
    dst[1] = (__bf16)(v.y * inv);
}

// ---------------------------------------------------------------------------
// B-fragment fetch from padded LDS tile (ISA 16-bit B layout, two b128 runs)
// ---------------------------------------------------------------------------
__device__ __forceinline__ v16bf ldB(const unsigned char* brow, int kk) {
    v8bf lo = *(const v8bf*)(brow + kk * 64);
    v8bf hi = *(const v8bf*)(brow + kk * 64 + 32);
    return __builtin_shufflevector(lo, hi, 0,1,2,3,4,5,6,7,8,9,10,11,12,13,14,15);
}

// ---------------------------------------------------------------------------
// Kernel 3: bf16 WMMA GEMM (scores = Q @ ff_norm^T) fused with top-1 argmax.
//   grid = (16 query-blocks of 128 rows, NSPLIT slabs of N)
//   block = 256 threads = 8 waves; wave w owns query rows qrow0..qrow0+15.
//   Double-buffered LDS: stage tile t+1 via GLOBAL_LOAD_ASYNC_TO_LDS_B128
//   (ASYNCcnt) while computing tile t; one barrier per iteration.
// ---------------------------------------------------------------------------
__global__ __launch_bounds__(256) void gemm_argmax(
        const __bf16* __restrict__ qb, const __bf16* __restrict__ ffb,
        unsigned long long* __restrict__ best) {
    __shared__ __align__(16) unsigned char lds[2 * BUF_SZ];

    const int tid   = threadIdx.x;
    const int lane  = tid & 31;
    const int wave  = tid >> 5;
    const int m     = lane & 15;
    const int khalf = lane >> 4;          // 0/1: ISA half-wave K split
    const int klo   = khalf * 8;          // first 8-half run offset (elements)
    const int qrow0 = blockIdx.x * 128 + wave * 16;

    // Per-thread staging geometry: 4 chunks of 16 B (1024 chunks = 16 KB tile)
    const int srow[4] = { (tid      ) >> 6, (tid + 256) >> 6,
                          (tid + 512) >> 6, (tid + 768) >> 6 };
    const int sch = tid & 63;

    // Preload full A panel (16 rows x 512 K) in ISA fragment layout: 128 VGPRs.
    v16bf a[16];
    {
        const __bf16* arow = qb + (size_t)(qrow0 + m) * D_F;
        #pragma unroll
        for (int kk = 0; kk < 16; ++kk) {
            const v8bf* p = (const v8bf*)(arow + kk * 32 + klo);
            v8bf lo = p[0];
            v8bf hi = p[2];               // +16 elements
            a[kk] = __builtin_shufflevector(lo, hi,
                        0,1,2,3,4,5,6,7,8,9,10,11,12,13,14,15);
        }
    }

    float bs[8];
    int   bi[8];
    #pragma unroll
    for (int r = 0; r < 8; ++r) { bs[r] = -3.402823e38f; bi[r] = 0; }

    const int t0 = blockIdx.y * TPB_T;

    // ---- prologue: stage tile 0 into buffer 0 ----
    {
        const int n0 = t0 * 16;
#if ASYNC_STAGE
        #pragma unroll
        for (int i = 0; i < 4; ++i) {
            const __bf16* g = ffb + (size_t)(n0 + srow[i]) * D_F + sch * 8;
            __builtin_amdgcn_global_load_async_to_lds_b128(
                (gv4i_p)(g),
                (lv4i_p)(lds + srow[i] * LDS_ROW + sch * 16),
                0, 0);
        }
        __builtin_amdgcn_s_wait_asynccnt(0);
#else
        #pragma unroll
        for (int i = 0; i < 4; ++i) {
            const uint4 s = *(const uint4*)(ffb + (size_t)(n0 + srow[i]) * D_F + sch * 8);
            *(uint4*)(lds + srow[i] * LDS_ROW + sch * 16) = s;
        }
#endif
        __syncthreads();
    }

    for (int t = 0; t < TPB_T; ++t) {
        const int cur = t & 1;
        const unsigned char* bufc = lds + cur * BUF_SZ;
        unsigned char*       bufn = lds + (1 - cur) * BUF_SZ;
        const bool have_next = (t + 1 < TPB_T);
        const int n0n = (t0 + t + 1) * 16;

#if ASYNC_STAGE
        // Kick off async staging of next tile directly into LDS (ASYNCcnt).
        if (have_next) {
            #pragma unroll
            for (int i = 0; i < 4; ++i) {
                const __bf16* g = ffb + (size_t)(n0n + srow[i]) * D_F + sch * 8;
                __builtin_amdgcn_global_load_async_to_lds_b128(
                    (gv4i_p)(g),
                    (lv4i_p)(bufn + srow[i] * LDS_ROW + sch * 16),
                    0, 0);
            }
        }
#else
        // Issue next tile's global loads early; park in VGPRs during compute.
        uint4 st[4];
        if (have_next) {
            #pragma unroll
            for (int i = 0; i < 4; ++i)
                st[i] = *(const uint4*)(ffb + (size_t)(n0n + srow[i]) * D_F + sch * 8);
        }
#endif

        // ---- 16 x WMMA K-loop, B fragment software-pipelined ----
        v8f c = {};
        const unsigned char* brow = bufc + (lane & 15) * LDS_ROW + klo * 2;
        v16bf bcur = ldB(brow, 0);
        #pragma unroll
        for (int kk = 0; kk < 15; ++kk) {
            v16bf bnext = ldB(brow, kk + 1);
            c = __builtin_amdgcn_wmma_f32_16x16x32_bf16(
                    false, a[kk], false, bcur, (short)0, c, false, false);
            bcur = bnext;
        }
        c = __builtin_amdgcn_wmma_f32_16x16x32_bf16(
                false, a[15], false, bcur, (short)0, c, false, false);

        // Per-lane running top-1: C VGPR r = row (r + khalf*8), col = lane&15.
        const int ncol = t0 * 16 + t * 16 + (lane & 15);
        #pragma unroll
        for (int r = 0; r < 8; ++r) {
            if (c[r] > bs[r]) { bs[r] = c[r]; bi[r] = ncol; }
        }

#if ASYNC_STAGE
        __builtin_amdgcn_s_wait_asynccnt(0);
#else
        if (have_next) {
            #pragma unroll
            for (int i = 0; i < 4; ++i)
                *(uint4*)(bufn + srow[i] * LDS_ROW + sch * 16) = st[i];
        }
#endif
        __syncthreads();
    }

    // Cross-lane (16-wide) argmax reduce + global merge.
    #pragma unroll
    for (int r = 0; r < 8; ++r) {
        unsigned int fb  = __float_as_uint(bs[r]);
        unsigned int key = (fb & 0x80000000u) ? ~fb : (fb | 0x80000000u);
        unsigned long long pk = ((unsigned long long)key << 32) | (unsigned int)bi[r];
        #pragma unroll
        for (int sh = 1; sh < 16; sh <<= 1) {
            unsigned int hi = (unsigned int)(pk >> 32);
            unsigned int lo = (unsigned int)(pk & 0xffffffffu);
            unsigned int ohi = __shfl_xor(hi, sh, 32);
            unsigned int olo = __shfl_xor(lo, sh, 32);
            unsigned long long o = ((unsigned long long)ohi << 32) | olo;
            if (o > pk) pk = o;
        }
        if ((lane & 15) == 0) {
            atomicMax(best + (qrow0 + r + khalf * 8), pk);   // global_atomic_max_u64
        }
    }
}

// ---------------------------------------------------------------------------
// Kernel 4: gather fp32 row of winner, scale by 1/norm, apply selector mask
// ---------------------------------------------------------------------------
__global__ void finalize(const float* __restrict__ ff,
                         const float* __restrict__ inv_norm,
                         const unsigned long long* __restrict__ best,
                         const unsigned char* __restrict__ sel,
                         float* __restrict__ out) {
    const int b = blockIdx.x;
    const int t = threadIdx.x;            // 256 threads, 2 floats each
    const unsigned long long pk = best[b];
    const int  idx = (int)(pk & 0xffffffffu);
    const bool s   = sel[b] != 0;
    const float inv = inv_norm[idx];
    const float2 v = ((const float2*)(ff + (size_t)idx * D_F))[t];
    float2 o;
    o.x = s ? v.x * inv : -1.0f;
    o.y = s ? v.y * inv : -1.0f;
    ((float2*)(out + (size_t)b * D_F))[t] = o;
}

// ---------------------------------------------------------------------------
extern "C" void kernel_launch(void* const* d_in, const int* in_sizes, int n_in,
                              void* d_out, int out_size, void* d_ws, size_t ws_size,
                              hipStream_t stream) {
    const float*         q   = (const float*)d_in[0];          // [B, D] fp32
    const float*         ff  = (const float*)d_in[1];          // [N, D] fp32
    const unsigned char* sel = (const unsigned char*)d_in[2];  // [B] bool (1 B)
    float*               out = (float*)d_out;                  // [B, D] fp32

    unsigned char* ws = (unsigned char*)d_ws;
    unsigned long long* best     = (unsigned long long*)ws;                    // 16 KB
    float*              inv_norm = (float*)(ws + 16384);                       // 400 KB
    __bf16*             qb       = (__bf16*)(ws + 16384 + 400128);             // 2 MB
    __bf16*             ffb      = (__bf16*)(ws + 16384 + 400128 + 2097152);   // 102.4 MB

    prep_q    <<<dim3((B_Q * D_F + 255) / 256), dim3(256), 0, stream>>>(q, qb, best);
    norm_rows <<<dim3(N_F),                      dim3(256), 0, stream>>>(ff, ffb, inv_norm);
    gemm_argmax<<<dim3(16, NSPLIT),              dim3(256), 0, stream>>>(qb, ffb, best);
    finalize  <<<dim3(B_Q),                      dim3(256), 0, stream>>>(ff, inv_norm, best, sel, out);
}
